// SwinTransformerBlock_40999757808183
// MI455X (gfx1250) — matrix operand order repaired and checked by reading, not compiled, not run on hardware
//
#include <hip/hip_runtime.h>

typedef __attribute__((ext_vector_type(16))) __bf16 v16bf;
typedef __attribute__((ext_vector_type(8)))  float  v8f;

// ---------- bf16 helpers (raw u16 storage, RNE convert) ----------
__device__ __forceinline__ unsigned short f2bf(float f) {
  unsigned u = __float_as_uint(f);
  unsigned r = u + 0x7FFFu + ((u >> 16) & 1u);
  return (unsigned short)(r >> 16);
}
__device__ __forceinline__ float bf2f(unsigned short h) {
  return __uint_as_float(((unsigned)h) << 16);
}

// Problem constants
#define MTOT 50176   // 16 * 56 * 56 tokens
#define CDIM 384
#define C3   1152
#define DFF  1536
#define NWIN 1024    // 16 * 8 * 8 windows
#define NTOK 49

// windowed row r -> natural token row (window-reverse + roll(+3)); same map
// as forward partition source (roll(-3) then partition).
__device__ __forceinline__ int rowmap(int r) {
  int w = r / 49, t = r - w * 49;
  int b = w >> 6, rem = w & 63, wh = rem >> 3, ww = rem & 7;
  int ii = t / 7, jj = t - ii * 7;
  int hs = wh * 7 + ii + 3; if (hs >= 56) hs -= 56;
  int ws = ww * 7 + jj + 3; if (ws >= 56) ws -= 56;
  return b * 3136 + hs * 56 + ws;
}

// ---------- weight pack: W [N x K] f32 (row-major) -> bf16 WMMA-B fragments ----------
// Fragment order: frag = kt*ntiles + nt; per-lane 8 dwords contiguous.
// Lane l: col n = nt*16 + (l&15); kbase = kt*32 + ((l&16)?16:0);
// dword j = pack(B[kbase+2j][n], B[kbase+2j+1][n]) with B[k][n] = W[n][k].
__global__ void pack_b_kernel(const float* __restrict__ W, unsigned* __restrict__ dst,
                              int K, int N) {
  int gid  = blockIdx.x * blockDim.x + threadIdx.x;
  int lane = gid & 31;
  int frag = gid >> 5;
  int ntiles = N >> 4;
  int kt = frag / ntiles;
  if (kt >= (K >> 5)) return;
  int nt = frag - kt * ntiles;
  int n  = nt * 16 + (lane & 15);
  int kb = kt * 32 + ((lane & 16) ? 16 : 0);
  const float* wrow = W + (size_t)n * K;
  unsigned* o = dst + ((size_t)frag * 32 + lane) * 8;
#pragma unroll
  for (int j = 0; j < 8; ++j) {
    unsigned lo = f2bf(wrow[kb + 2 * j]);
    unsigned hi = f2bf(wrow[kb + 2 * j + 1]);
    o[j] = lo | (hi << 16);
  }
}

// ---------- LayerNorm (wave per row); windowed=1 fuses shift+partition ----------
__global__ void ln_kernel(const float* __restrict__ src, const float* __restrict__ gam,
                          const float* __restrict__ bet, unsigned short* __restrict__ dst,
                          int windowed) {
  int wave = threadIdx.x >> 5;
  int lane = threadIdx.x & 31;
  int r = blockIdx.x * 8 + wave;            // destination row
  int srow = windowed ? rowmap(r) : r;
  const float* s = src + (size_t)srow * CDIM;
  float v[12];
  float s1 = 0.f, s2 = 0.f;
#pragma unroll
  for (int k = 0; k < 12; ++k) {
    float f = s[k * 32 + lane];
    v[k] = f; s1 += f; s2 += f * f;
  }
#pragma unroll
  for (int off = 16; off; off >>= 1) {
    s1 += __shfl_xor(s1, off, 32);
    s2 += __shfl_xor(s2, off, 32);
  }
  float mean = s1 * (1.f / CDIM);
  float var  = s2 * (1.f / CDIM) - mean * mean;
  float rstd = rsqrtf(var + 1e-5f);
  unsigned short* o = dst + (size_t)r * CDIM;
#pragma unroll
  for (int k = 0; k < 12; ++k) {
    int c = k * 32 + lane;
    o[c] = f2bf((v[k] - mean) * rstd * gam[c] + bet[c]);
  }
}

// ---------- WMMA GEMM: C[M,N] = A[M,K](bf16) * Bpacked + bias -------------------
// 256 threads = 8 waves (4x2); each wave computes a 32x64 tile (2x4 WMMA accums),
// block tile 128x128. 8 wmma per 12 global_load_b128 per K-step of 32.
// ~140 live VGPRs -> no spills (64x64 spilled, 32x32 under-utilized).
// MODE 0: bf16 out            MODE 1: bf16 GELU out
// MODE 2: f32 scatter (window-reverse + unshift) + resid add (proj)
// MODE 3: f32 out = resid + acc (mlp2, natural rows)
template <int MODE>
__global__ void gemm_wmma(const unsigned short* __restrict__ A,
                          const uint4* __restrict__ Bp,
                          const float* __restrict__ bias,
                          int N, int K,
                          unsigned short* __restrict__ outb,
                          float* __restrict__ outf,
                          const float* __restrict__ resid) {
  int lane = threadIdx.x & 31;
  int wv   = threadIdx.x >> 5;
  int wr = wv >> 1, wc = wv & 1;
  int m0 = blockIdx.y * 128 + wr * 32;
  int n0 = blockIdx.x * 128 + wc * 64;
  int lm = lane & 15;
  int kbA = (lane & 16) ? 16 : 0;           // A layout: hi lanes hold K+16..31
  int ntiles = N >> 4;
  int nt0 = n0 >> 4;

  v8f acc[2][4] = {};
  const unsigned short* arow[2];
  arow[0] = A + (size_t)(m0 + lm) * K + kbA;
  arow[1] = A + (size_t)(m0 + 16 + lm) * K + kbA;

  union Frag { uint4 q[2]; v16bf v; };

  int ksteps = K >> 5;
  for (int kt = 0; kt < ksteps; ++kt) {
    Frag a[2], b[4];
#pragma unroll
    for (int i = 0; i < 2; ++i) {
      const unsigned short* ar = arow[i] + kt * 32;
      a[i].q[0] = *(const uint4*)ar;          // K = kbA + 0..7
      a[i].q[1] = *(const uint4*)(ar + 8);    // K = kbA + 8..15
    }
#pragma unroll
    for (int j = 0; j < 4; ++j) {
      const uint4* pb = Bp + ((size_t)(kt * ntiles + nt0 + j) * 32 + lane) * 2;
      b[j].q[0] = pb[0];
      b[j].q[1] = pb[1];
    }
#pragma unroll
    for (int i = 0; i < 2; ++i)
#pragma unroll
      for (int j = 0; j < 4; ++j)
        acc[i][j] = __builtin_amdgcn_wmma_f32_16x16x32_bf16(
            false, a[i].v, false, b[j].v, (short)0, acc[i][j], false, false);
  }

  int rbase = (lane & 16) ? 8 : 0;
  float bv[4];
#pragma unroll
  for (int j = 0; j < 4; ++j) bv[j] = bias[n0 + j * 16 + lm];

#pragma unroll
  for (int i = 0; i < 2; ++i) {
#pragma unroll
    for (int r = 0; r < 8; ++r) {
      int mm = m0 + i * 16 + rbase + r;
      size_t rowbase;
      if (MODE == 2) rowbase = (size_t)rowmap(mm) * CDIM;
      else           rowbase = (size_t)mm * N;
#pragma unroll
      for (int j = 0; j < 4; ++j) {
        int nn = n0 + j * 16 + lm;
        float val = acc[i][j][r] + bv[j];
        if (MODE == 0) {
          outb[rowbase + nn] = f2bf(val);
        } else if (MODE == 1) {
          float g = 0.5f * val * (1.0f + erff(val * 0.70710678118654752f));
          outb[rowbase + nn] = f2bf(g);
        } else {  // MODE 2 / 3: f32 residual add
          size_t idx = rowbase + nn;
          outf[idx] = resid[idx] + val;
        }
      }
    }
  }
}

// ---------- per-(window, head) attention: softmax(q k^T * s + bias) v -----------
__global__ void attn_kernel(const unsigned short* __restrict__ qkv,
                            const float* __restrict__ rpb,
                            unsigned short* __restrict__ o) {
  __shared__ float ks[NTOK * 32];
  __shared__ float vs[NTOK * 32];
  int w = blockIdx.x, head = blockIdx.y;
  int tid = threadIdx.x;                    // 64 threads
  for (int idx = tid; idx < NTOK * 32; idx += 64) {
    int m = idx >> 5, d = idx & 31;
    size_t base = (size_t)(w * NTOK + m) * C3 + head * 32 + d;
    ks[idx] = bf2f(qkv[base + CDIM]);       // k
    vs[idx] = bf2f(qkv[base + 2 * CDIM]);   // v
  }
  __syncthreads();
  int t = tid;
  if (t < NTOK) {
    float q[32];
    size_t qb = (size_t)(w * NTOK + t) * C3 + head * 32;
#pragma unroll
    for (int d = 0; d < 32; ++d) q[d] = bf2f(qkv[qb + d]);
    int it = t / 7, jt = t - it * 7;
    float s[NTOK];
    float mx = -1e30f;
    for (int m = 0; m < NTOK; ++m) {
      float acc = 0.f;
#pragma unroll
      for (int d = 0; d < 32; ++d) acc += q[d] * ks[m * 32 + d];
      int im = m / 7, jm = m - im * 7;
      int ridx = (it - im + 6) * 13 + (jt - jm + 6);
      acc = acc * 0.17677669529663687f + rpb[ridx * 12 + head];
      s[m] = acc;
      mx = fmaxf(mx, acc);
    }
    float sum = 0.f;
    for (int m = 0; m < NTOK; ++m) { s[m] = expf(s[m] - mx); sum += s[m]; }
    float inv = 1.0f / sum;
    float ov[32];
#pragma unroll
    for (int d = 0; d < 32; ++d) ov[d] = 0.f;
    for (int m = 0; m < NTOK; ++m) {
      float p = s[m] * inv;
#pragma unroll
      for (int d = 0; d < 32; ++d) ov[d] += p * vs[m * 32 + d];
    }
    unsigned short* op = o + (size_t)(w * NTOK + t) * CDIM + head * 32;
#pragma unroll
    for (int d = 0; d < 32; ++d) op[d] = f2bf(ov[d]);
  }
}

// ---------------- launcher ----------------
extern "C" void kernel_launch(void* const* d_in, const int* in_sizes, int n_in,
                              void* d_out, int out_size, void* d_ws, size_t ws_size,
                              hipStream_t stream) {
  (void)in_sizes; (void)n_in; (void)out_size; (void)ws_size;
  const float* x     = (const float*)d_in[0];
  const float* n1g   = (const float*)d_in[1];
  const float* n1b   = (const float*)d_in[2];
  const float* qkvw  = (const float*)d_in[3];
  const float* qkvb  = (const float*)d_in[4];
  const float* projw = (const float*)d_in[5];
  const float* projb = (const float*)d_in[6];
  const float* rpb   = (const float*)d_in[7];
  const float* n2g   = (const float*)d_in[8];
  const float* n2b   = (const float*)d_in[9];
  const float* w1    = (const float*)d_in[10];
  const float* b1    = (const float*)d_in[11];
  const float* w2    = (const float*)d_in[12];
  const float* b2    = (const float*)d_in[13];
  float* out = (float*)d_out;

  char* ws = (char*)d_ws;
  // packed weights (bf16)
  unsigned* p_qkvw = (unsigned*)(ws + 0);           //   884,736 B
  unsigned* p_projw= (unsigned*)(ws + 884736);      //   294,912 B
  unsigned* p_w1   = (unsigned*)(ws + 1179648);     // 1,179,648 B
  unsigned* p_w2   = (unsigned*)(ws + 2359296);     // 1,179,648 B
  // activations
  float*          y1  = (float*)(ws + (size_t)4194304);              // 77,070,336 B f32
  unsigned short* med = (unsigned short*)(ws + (size_t)81264640);    // 38,535,168 B bf16 (xw / o / h2)
  unsigned short* big = (unsigned short*)(ws + (size_t)119799808);   // 154,140,672 B bf16 (qkv / gelu)

  // 1) pack weights into WMMA-B fragment order
  pack_b_kernel<<<108, 256, 0, stream>>>(qkvw,  p_qkvw,  CDIM, C3);
  pack_b_kernel<<< 36, 256, 0, stream>>>(projw, p_projw, CDIM, CDIM);
  pack_b_kernel<<<144, 256, 0, stream>>>(w1,    p_w1,    CDIM, DFF);
  pack_b_kernel<<<144, 256, 0, stream>>>(w2,    p_w2,    DFF,  CDIM);

  // 2) LN1 + shift + window partition -> med (bf16, windowed rows)
  ln_kernel<<<MTOT / 8, 256, 0, stream>>>(x, n1g, n1b, med, 1);

  // 3) QKV GEMM -> big (bf16)
  gemm_wmma<0><<<dim3(C3 / 128, MTOT / 128), 256, 0, stream>>>(
      med, (const uint4*)p_qkvw, qkvb, C3, CDIM, big, nullptr, nullptr);

  // 4) window attention -> med (bf16, heads concatenated)
  attn_kernel<<<dim3(NWIN, 12), 64, 0, stream>>>(big, rpb, med);

  // 5) proj GEMM + window-reverse/unshift scatter + residual -> y1 (f32, natural rows)
  gemm_wmma<2><<<dim3(CDIM / 128, MTOT / 128), 256, 0, stream>>>(
      med, (const uint4*)p_projw, projb, CDIM, CDIM, nullptr, y1, x);

  // 6) LN2 -> med (bf16)
  ln_kernel<<<MTOT / 8, 256, 0, stream>>>(y1, n2g, n2b, med, 0);

  // 7) MLP up-proj + GELU -> big (bf16)
  gemm_wmma<1><<<dim3(DFF / 128, MTOT / 128), 256, 0, stream>>>(
      med, (const uint4*)p_w1, b1, DFF, CDIM, big, nullptr, nullptr);

  // 8) MLP down-proj + residual -> d_out (f32)
  gemm_wmma<3><<<dim3(CDIM / 128, MTOT / 128), 256, 0, stream>>>(
      big, (const uint4*)p_w2, b2, CDIM, DFF, nullptr, out, y1);
}
